// AdaptiveBlockSelector_41171556500245
// MI455X (gfx1250) — matrix operand order repaired and checked by reading, not compiled, not run on hardware
//
#include <hip/hip_runtime.h>
#include <hip/hip_bf16.h>

// ---------------------------------------------------------------------------
// AdaptiveBlockSelector for MI455X (gfx1250, wave32, WMMA)
//
// scores = (l2norm(q) @ l2norm(k)^T) / TEMP + mask ; out = one-hot(top8(scores))
// B=4, Qb=Bb=4096, C=128.  Output (256 MB) dominates -> fuse GEMM + top-k,
// NT-store the one-hot mask, run the GEMM on v_wmma_f32_16x16x32_f16.
// Round 2: software-pipeline the B-fragment loads (distinct regs for all 4
// k-steps so all 8 b128 loads clause up before the first WMMA), vectorize the
// LDS top-k scan reads to ds_load_b128.
// ---------------------------------------------------------------------------

typedef __attribute__((ext_vector_type(16))) _Float16 v16h;
typedef __attribute__((ext_vector_type(8)))  _Float16 v8h;
typedef __attribute__((ext_vector_type(8)))  float    v8f;
typedef __attribute__((ext_vector_type(4)))  float    v4f;
typedef __attribute__((ext_vector_type(4)))  _Float16 v4h;

#define B_   4
#define QB_  4096
#define BB_  4096
#define C_   128
#define KTOP 8

__device__ __forceinline__ v16h cat8(v8h lo, v8h hi) {
  return __builtin_shufflevector(lo, hi, 0,1,2,3,4,5,6,7,8,9,10,11,12,13,14,15);
}

// sorted-descending top-8 insertion; fully unrolled -> stays in VGPRs
__device__ __forceinline__ void topk_insert(float (&vals)[KTOP], int (&idxs)[KTOP],
                                            float v, int id) {
  if (v > vals[KTOP - 1]) {
    vals[KTOP - 1] = v; idxs[KTOP - 1] = id;
#pragma unroll
    for (int s = KTOP - 1; s > 0; --s) {
      if (vals[s] > vals[s - 1]) {
        float tv = vals[s]; vals[s] = vals[s - 1]; vals[s - 1] = tv;
        int   ti = idxs[s]; idxs[s] = idxs[s - 1]; idxs[s - 1] = ti;
      }
    }
  }
}

// --------------------- kernel 1: row L2-normalize -> f16 -------------------
// one wave per 128-float row; wave32 xor-shuffle reduction
__global__ __launch_bounds__(256) void l2norm_f16_kernel(const float* __restrict__ in,
                                                         _Float16* __restrict__ out,
                                                         int nrows) {
  int row  = blockIdx.x * 8 + (threadIdx.x >> 5);
  int lane = threadIdx.x & 31;
  if (row >= nrows) return;
  const float* p = in + (size_t)row * C_ + lane * 4;
  v4f v = *(const v4f*)p;
  float ss = v[0]*v[0] + v[1]*v[1] + v[2]*v[2] + v[3]*v[3];
#pragma unroll
  for (int off = 16; off > 0; off >>= 1) ss += __shfl_xor(ss, off, 32);
  float inv = 1.0f / fmaxf(sqrtf(ss), 1e-12f);   // matches F.normalize eps
  v4h o;
  o[0] = (_Float16)(v[0] * inv); o[1] = (_Float16)(v[1] * inv);
  o[2] = (_Float16)(v[2] * inv); o[3] = (_Float16)(v[3] * inv);
  *(v4h*)(out + (size_t)row * C_ + lane * 4) = o;
}

// ------------- kernel 2: fused WMMA GEMM + top-8 + one-hot write -----------
// 1 workgroup (8 waves) = 1 (batch, 16-query) tile; wave w owns keys w*16..w*16+15
// of each 128-key iteration.
__global__ __launch_bounds__(256) void topk_select_kernel(const _Float16* __restrict__ qn,
                                                          const _Float16* __restrict__ kn,
                                                          const float* __restrict__ mask,
                                                          float* __restrict__ out) {
  __shared__ float sScores[16][132];   // 16 q-rows x 128 keys (+pad, 528B rows)
  __shared__ float sCVal[16][128];     // per-(row,segment) top-8 candidates
  __shared__ int   sCIdx[16][128];
  __shared__ int   sTopIdx[16][KTOP];

  const int tid   = threadIdx.x;
  const int lane  = tid & 31;
  const int wave  = tid >> 5;
  const int b     = blockIdx.x >> 8;          // QB_/16 == 256 tiles per batch
  const int qrow0 = (blockIdx.x & 255) << 4;

  // ---- A fragments (16x32 f16 per K-step), loaded once, reused 32x ----
  // lane<16: row=lane,    K-halves {0..7,16..23}  -> base offset 0
  // lane>=16: row=lane-16, K-halves {8..15,24..31} -> base offset 8
  const int am = lane & 15;
  const _Float16* qrow =
      qn + ((size_t)b * QB_ + qrow0 + am) * C_ + ((lane & 16) ? 8 : 0);
  v16h afrag[4];
#pragma unroll
  for (int s = 0; s < 4; ++s) {
    v8h lo = *(const v8h*)(qrow + s * 32);
    v8h hi = *(const v8h*)(qrow + s * 32 + 16);
    afrag[s] = cat8(lo, hi);
  }

  float vals[KTOP]; int idxs[KTOP];
#pragma unroll
  for (int j = 0; j < KTOP; ++j) { vals[j] = -__builtin_inff(); idxs[j] = 0; }

  const int myRow = tid >> 4;     // query row scanned by this thread
  const int mySeg = tid & 15;     // 8-key segment within the 128-key tile
  const float invTemp = 1.0f / (0.05f + 1e-8f);

  for (int keyBase = 0; keyBase < BB_; keyBase += 128) {
    const int key = keyBase + wave * 16 + (lane & 15);
    // B fragment: lane<16 holds K {0..7,16..23}, lane>=16 holds K {8..15,24..31} of its key column
    const _Float16* krow =
        kn + ((size_t)b * BB_ + key) * C_ + ((lane & 16) ? 8 : 0);
    __builtin_prefetch(krow + (size_t)128 * C_, 0, 1);   // next key tile

    // issue mask dword + all 8 b128 B-loads before any WMMA so the
    // scheduler can clause them and retire WMMAs on partial loadcnt waits
    const float mval = mask[(size_t)b * BB_ + key];
    v16h bfrag[4];
#pragma unroll
    for (int s = 0; s < 4; ++s) {
      v8h lo = *(const v8h*)(krow + s * 32);
      v8h hi = *(const v8h*)(krow + s * 32 + 16);
      bfrag[s] = cat8(lo, hi);
    }

    v8f acc = {};
#pragma unroll
    for (int s = 0; s < 4; ++s)
      acc = __builtin_amdgcn_wmma_f32_16x16x32_f16(
          false, afrag[s], false, bfrag[s], (short)0, acc, false, false);

    // scale + additive mask, spill 16x16 sub-tile to LDS
    const int rbase = (lane & 16) ? 8 : 0;   // acc VGPR i -> row rbase+i
    const int kl    = wave * 16 + (lane & 15);
#pragma unroll
    for (int i = 0; i < 8; ++i)
      sScores[rbase + i][kl] = acc[i] * invTemp + mval;
    __syncthreads();

    // 256 threads = 16 rows x 16 segments: each keeps a running top-8.
    // 8 contiguous, 16B-aligned floats -> two ds_load_b128
    {
      v4f s0 = *(const v4f*)&sScores[myRow][mySeg * 8];
      v4f s1 = *(const v4f*)&sScores[myRow][mySeg * 8 + 4];
      const int gbase = keyBase + mySeg * 8;
#pragma unroll
      for (int j = 0; j < 4; ++j) topk_insert(vals, idxs, s0[j], gbase + j);
#pragma unroll
      for (int j = 0; j < 4; ++j) topk_insert(vals, idxs, s1[j], gbase + 4 + j);
    }
    __syncthreads();
  }

  // ---- merge 16 partial top-8 lists per row ----
#pragma unroll
  for (int j = 0; j < KTOP; ++j) {
    sCVal[myRow][mySeg * 8 + j] = vals[j];
    sCIdx[myRow][mySeg * 8 + j] = idxs[j];
  }
  __syncthreads();
  if (tid < 16) {
    float tv[KTOP]; int ti[KTOP];
#pragma unroll
    for (int j = 0; j < KTOP; ++j) { tv[j] = -__builtin_inff(); ti[j] = 0; }
    for (int c = 0; c < 128; ++c)
      topk_insert(tv, ti, sCVal[tid][c], sCIdx[tid][c]);
#pragma unroll
    for (int j = 0; j < KTOP; ++j) sTopIdx[tid][j] = ti[j];
  }
  __syncthreads();

  // ---- one-hot output: thread t owns columns [t*16, t*16+16) of each row ----
  // bitmask instead of dynamically-indexed array -> no scratch spills
  for (int r = 0; r < 16; ++r) {
    unsigned msk = 0;
#pragma unroll
    for (int j = 0; j < KTOP; ++j) {
      int id = sTopIdx[r][j];
      if ((id >> 4) == tid) msk |= 1u << (id & 15);
    }
    float* op = out + ((size_t)b * QB_ + qrow0 + r) * BB_ + tid * 16;
#pragma unroll
    for (int i = 0; i < 4; ++i) {
      v4f st = { (msk >> (4*i + 0)) & 1 ? 1.0f : 0.0f,
                 (msk >> (4*i + 1)) & 1 ? 1.0f : 0.0f,
                 (msk >> (4*i + 2)) & 1 ? 1.0f : 0.0f,
                 (msk >> (4*i + 3)) & 1 ? 1.0f : 0.0f };
      __builtin_nontemporal_store(st, (v4f*)(op + 4 * i));  // 256MB, bypass L2 reuse
    }
  }
}

extern "C" void kernel_launch(void* const* d_in, const int* in_sizes, int n_in,
                              void* d_out, int out_size, void* d_ws, size_t ws_size,
                              hipStream_t stream) {
  const float* q    = (const float*)d_in[0];   // (4,4096,128) f32
  const float* k    = (const float*)d_in[1];   // (4,4096,128) f32
  const float* mask = (const float*)d_in[2];   // (4,4096)     f32
  float* out = (float*)d_out;                  // (4,4096,4096) f32

  _Float16* qn = (_Float16*)d_ws;                         // 4 MB
  _Float16* kn = qn + (size_t)B_ * QB_ * C_;              // +4 MB

  const int nrows = B_ * QB_;                  // 16384 rows each
  l2norm_f16_kernel<<<nrows / 8, 256, 0, stream>>>(q, qn, nrows);
  l2norm_f16_kernel<<<nrows / 8, 256, 0, stream>>>(k, kn, nrows);

  // 4 batches x 256 query-tiles
  topk_select_kernel<<<B_ * (QB_ / 16), 256, 0, stream>>>(qn, kn, mask, out);
}
